// SW_MSA_21225728377066
// MI455X (gfx1250) — matrix operand order repaired and checked by reading, not compiled, not run on hardware
//
#include <hip/hip_runtime.h>

typedef __attribute__((ext_vector_type(16))) __bf16 v16bf;
typedef __attribute__((ext_vector_type(8)))  float  v8f;

#define NTOK 49
#define NPAD 64
#define CDIM 192
#define HEADS 6
#define HD 32
#define QKV_ROWS 576
#define IMGW 56
#define IMGTOK 3136

__device__ __forceinline__ unsigned short f32_to_bf16(float f) {
  unsigned int u = __float_as_uint(f);
  unsigned int r = 0x7FFFu + ((u >> 16) & 1u);
  return (unsigned short)((u + r) >> 16);
}

__device__ __forceinline__ v8f vzero8() {
  v8f a;
#pragma unroll
  for (int i = 0; i < 8; ++i) a[i] = 0.0f;
  return a;
}

__device__ __forceinline__ v8f wmma_bf16(v16bf a, v16bf b, v8f c) {
  // D = A(16x32 bf16) * B(32x16 bf16) + C(16x16 f32)
  return __builtin_amdgcn_wmma_f32_16x16x32_bf16(false, a, false, b, (short)0, c,
                                                 false, false);
}

// B-fragment (32x16): lane l holds N=l%16, K = (l/16)*16 + j (16 consecutive elems).
// Caller passes p = base + row(l%16)*ld + colbase + (l/16)*16.
__device__ __forceinline__ v16bf load_b_frag(const unsigned short* p) {
  union { v16bf v; uint4 q[2]; } u;
  u.q[0] = *(const uint4*)(p);
  u.q[1] = *(const uint4*)(p + 8);
  return u.v;
}

// A-fragment (16x32): lane l holds M=row0+l%16, K = (l/16)*16 + j (16 consecutive elems).
__device__ __forceinline__ v16bf load_a_frag(const unsigned short* base, int ld,
                                             int row, int col0, int lhalf) {
  const unsigned short* p = base + row * ld + col0 + lhalf * 16;
  union { v16bf v; uint4 q[2]; } u;
  u.q[0] = *(const uint4*)(p);        // K = col0 + lhalf*16 .. +7
  u.q[1] = *(const uint4*)(p + 8);    // K = col0 + lhalf*16 + 8 .. +15
  return u.v;
}

// ---------------- stage 0: weights f32 -> bf16 into workspace ----------------
__global__ void prepack_weights(const float* __restrict__ qkv_w,
                                const float* __restrict__ proj_w,
                                unsigned short* __restrict__ wbf) {
  int i = blockIdx.x * 256 + threadIdx.x;
  const int nq = QKV_ROWS * CDIM;          // 110592
  const int np = CDIM * CDIM;              // 36864
  if (i < nq)            wbf[i] = f32_to_bf16(qkv_w[i]);
  else if (i < nq + np)  wbf[i] = f32_to_bf16(proj_w[i - nq]);
}

// ---------------- fused Swin W-MSA: one workgroup per window ----------------
// LDS ~153.5KB -> 2 workgroups per 320KB WGP.
__global__ __launch_bounds__(256)
void swin_wmsa_kernel(const float* __restrict__ x,
                      const unsigned short* __restrict__ qkv_wb,   // bf16 576x192
                      const float* __restrict__ qkv_b,
                      const unsigned short* __restrict__ proj_wb,  // bf16 192x192
                      const float* __restrict__ proj_b,
                      const float* __restrict__ amask,             // 64x49x49 f32
                      float* __restrict__ out) {
  extern __shared__ char smem[];
  unsigned short* XsOb = (unsigned short*)smem;               // Xs (stage1) / Ob (4,5)
  unsigned short* Qs = XsOb + NPAD * CDIM;                    // 64x192 (pre-scaled)
  unsigned short* Ks = Qs + NPAD * CDIM;                      // 64x192
  unsigned short* Vt = Ks + NPAD * CDIM;                      // 192x64 (dim-major)
  unsigned short* aP = Vt + CDIM * NPAD;                      // 6x64x64 probs (bf16)
  float*          Ms = (float*)(aP + HEADS * NPAD * NPAD);    // 49x49 mask

  const int tid   = threadIdx.x;
  const int wave  = tid >> 5;
  const int lane  = tid & 31;
  const int l16   = lane & 15;
  const int lhalf = lane >> 4;

  const int wid  = blockIdx.x;
  const int bimg = wid >> 6;
  const int wli  = wid & 63;       // window index inside image -> mask row
  const int hw   = wli >> 3, ww = wli & 7;

  // ---- load shifted window -> Xs (bf16), zero-pad tokens 49..63; mask -> Ms ----
  unsigned short* Xs = XsOb;
  for (int idx = tid; idx < NPAD * CDIM; idx += 256) {
    int t = idx / CDIM, ch = idx % CDIM;
    unsigned short v = 0;
    if (t < NTOK) {
      int r = t / 7, c = t % 7;
      int sh = (hw * 7 + r + 3) % IMGW;   // roll(-3): src = pos + 3 mod 56
      int sw = (ww * 7 + c + 3) % IMGW;
      v = f32_to_bf16(x[((size_t)bimg * IMGTOK + sh * IMGW + sw) * CDIM + ch]);
    }
    Xs[idx] = v;
  }
  {
    const float* mwin = amask + (size_t)wli * (NTOK * NTOK);
    for (int idx = tid; idx < NTOK * NTOK; idx += 256) Ms[idx] = mwin[idx];
  }
  __syncthreads();

  // ---- stage 1: qkv = Xs @ qkv_w^T + b ; split into Qs (scaled), Ks, Vt ----
  const float qscale = 0.17677669529663687f;  // 32^-0.5
  for (int ot = wave; ot < QKV_ROWS / 16; ot += 8) {
    const int o0 = ot * 16;
    v8f acc[4];
#pragma unroll
    for (int m = 0; m < 4; ++m) acc[m] = vzero8();
#pragma unroll
    for (int ks = 0; ks < 6; ++ks) {
      const int c0 = ks * 32;
      v16bf bf = load_b_frag(qkv_wb + (size_t)(o0 + l16) * CDIM + c0 + lhalf * 16);
#pragma unroll
      for (int m = 0; m < 4; ++m) {
        v16bf af = load_a_frag(Xs, CDIM, m * 16 + l16, c0, lhalf);
        acc[m] = wmma_bf16(af, bf, acc[m]);
      }
    }
    const float bias = qkv_b[o0 + l16];
    const bool isq = (o0 < CDIM);
#pragma unroll
    for (int m = 0; m < 4; ++m) {
#pragma unroll
      for (int r = 0; r < 8; ++r) {
        float vr = acc[m][r] + bias;
        if (isq) vr *= qscale;
        const int tok = m * 16 + r + 8 * lhalf;
        const unsigned short hb = f32_to_bf16(vr);
        if (o0 < CDIM)            Qs[tok * CDIM + o0 + l16] = hb;
        else if (o0 < 2 * CDIM)   Ks[tok * CDIM + (o0 - CDIM) + l16] = hb;
        else                      Vt[((o0 - 2 * CDIM) + l16) * NPAD + tok] = hb;
      }
    }
  }
  __syncthreads();

  // ---- stages 2-4 (per-head, wave h): logits -> register softmax -> P@V ----
  unsigned short* Ob = XsOb;  // Xs is dead; reuse as proj input buffer
  if (wave < HEADS) {
    const int h = wave;

    // logits: 4x4 D tiles, K = 32 (= head dim) in one WMMA each
    v8f lg[4][4];
#pragma unroll
    for (int mt = 0; mt < 4; ++mt) {
      v16bf qf = load_a_frag(Qs, CDIM, mt * 16 + l16, h * HD, lhalf);
#pragma unroll
      for (int nt = 0; nt < 4; ++nt) {
        v16bf kf = load_b_frag(Ks + (nt * 16 + l16) * CDIM + h * HD + lhalf * 16);
        lg[mt][nt] = wmma_bf16(qf, kf, vzero8());
      }
    }

    // softmax fully in registers. Row m = mt*16 + r + 8*lhalf lives in one
    // 16-lane half across the 4 n-tiles -> xor-shuffle masks 1/2/4/8 reduce it.
    unsigned short* Ph = aP + (size_t)h * NPAD * NPAD;
#pragma unroll
    for (int mt = 0; mt < 4; ++mt) {
#pragma unroll
      for (int r = 0; r < 8; ++r) {
        const int m = mt * 16 + r + 8 * lhalf;
        float v[4];
#pragma unroll
        for (int nt = 0; nt < 4; ++nt) {
          const int n = nt * 16 + l16;
          float t = -1e30f;
          if (m < NTOK && n < NTOK) t = lg[mt][nt][r] + Ms[m * NTOK + n];
          v[nt] = t;
        }
        float mx = fmaxf(fmaxf(v[0], v[1]), fmaxf(v[2], v[3]));
#pragma unroll
        for (int xm = 1; xm < 16; xm <<= 1) mx = fmaxf(mx, __shfl_xor(mx, xm, 32));
        float e[4], s = 0.0f;
#pragma unroll
        for (int nt = 0; nt < 4; ++nt) { e[nt] = __expf(v[nt] - mx); s += e[nt]; }
#pragma unroll
        for (int xm = 1; xm < 16; xm <<= 1) s += __shfl_xor(s, xm, 32);
        const float inv = 1.0f / s;
#pragma unroll
        for (int nt = 0; nt < 4; ++nt) {
          const int n = nt * 16 + l16;
          Ph[m * NPAD + n] = (m < NTOK && n < NTOK) ? f32_to_bf16(e[nt] * inv)
                                                    : (unsigned short)0;
        }
      }
    }

    // O_head = P @ V : M=64, N=32 (2 tiles), K=64 (2 steps). P produced and
    // consumed by this same wave -> no barrier needed in between.
    v8f acc[4][2];
#pragma unroll
    for (int mt = 0; mt < 4; ++mt)
#pragma unroll
      for (int nt = 0; nt < 2; ++nt) acc[mt][nt] = vzero8();
#pragma unroll
    for (int kt = 0; kt < 2; ++kt) {
      const int k0 = kt * 32;
      v16bf bfr[2];
#pragma unroll
      for (int nt = 0; nt < 2; ++nt)
        bfr[nt] = load_b_frag(Vt + (h * HD + nt * 16 + l16) * NPAD + k0 + lhalf * 16);
#pragma unroll
      for (int mt = 0; mt < 4; ++mt) {
        v16bf af = load_a_frag(Ph, NPAD, mt * 16 + l16, k0, lhalf);
#pragma unroll
        for (int nt = 0; nt < 2; ++nt) acc[mt][nt] = wmma_bf16(af, bfr[nt], acc[mt][nt]);
      }
    }
#pragma unroll
    for (int mt = 0; mt < 4; ++mt)
#pragma unroll
      for (int nt = 0; nt < 2; ++nt)
#pragma unroll
        for (int r = 0; r < 8; ++r) {
          const int tok = mt * 16 + r + 8 * lhalf;
          Ob[tok * CDIM + h * HD + nt * 16 + l16] = f32_to_bf16(acc[mt][nt][r]);
        }
  }
  __syncthreads();

  // ---- stage 5: out = Ob @ proj_w^T + b, scatter with reverse shift ----
  for (int ot = wave; ot < CDIM / 16; ot += 8) {
    const int o0 = ot * 16;
    v8f acc[4];
#pragma unroll
    for (int m = 0; m < 4; ++m) acc[m] = vzero8();
#pragma unroll
    for (int ks = 0; ks < 6; ++ks) {
      const int c0 = ks * 32;
      v16bf bf = load_b_frag(proj_wb + (size_t)(o0 + l16) * CDIM + c0 + lhalf * 16);
#pragma unroll
      for (int m = 0; m < 4; ++m) {
        v16bf af = load_a_frag(Ob, CDIM, m * 16 + l16, c0, lhalf);
        acc[m] = wmma_bf16(af, bf, acc[m]);
      }
    }
    const float bias = proj_b[o0 + l16];
#pragma unroll
    for (int m = 0; m < 4; ++m)
#pragma unroll
      for (int r = 0; r < 8; ++r) {
        const int tok = m * 16 + r + 8 * lhalf;
        if (tok < NTOK) {
          const int rr = tok / 7, cc = tok % 7;
          const int dh = (hw * 7 + rr + 3) % IMGW;  // reverse roll(+3) == same map
          const int dw = (ww * 7 + cc + 3) % IMGW;
          out[((size_t)bimg * IMGTOK + dh * IMGW + dw) * CDIM + o0 + l16] =
              acc[m][r] + bias;
        }
      }
  }
}

extern "C" void kernel_launch(void* const* d_in, const int* in_sizes, int n_in,
                              void* d_out, int out_size, void* d_ws, size_t ws_size,
                              hipStream_t stream) {
  const float* x      = (const float*)d_in[0];
  const float* qkv_w  = (const float*)d_in[1];
  const float* qkv_b  = (const float*)d_in[2];
  const float* proj_w = (const float*)d_in[3];
  const float* proj_b = (const float*)d_in[4];
  const float* amask  = (const float*)d_in[5];
  float* out = (float*)d_out;

  unsigned short* wbf = (unsigned short*)d_ws;  // [0,110592): qkv bf16, then proj bf16
  const int nq = QKV_ROWS * CDIM, np = CDIM * CDIM;
  prepack_weights<<<(nq + np + 255) / 256, 256, 0, stream>>>(qkv_w, proj_w, wbf);

  const size_t shmem =
      (size_t)(NPAD * CDIM) * sizeof(unsigned short)                    // Xs / Ob
      + (size_t)(2 * NPAD * CDIM + CDIM * NPAD) * sizeof(unsigned short) // Qs,Ks,Vt
      + (size_t)HEADS * NPAD * NPAD * sizeof(unsigned short)            // probs
      + (size_t)NTOK * NTOK * sizeof(float);                            // mask
  // 4096 windows (B=64 x 64 windows), 8 waves each; ~153.5KB LDS -> 2 WGs/WGP
  swin_wmsa_kernel<<<4096, 256, shmem, stream>>>(x, wbf, qkv_b, wbf + nq, proj_b,
                                                 amask, out);
}
